// GATe_85899345975
// MI455X (gfx1250) — matrix operand order repaired and checked by reading, not compiled, not run on hardware
//
#include <hip/hip_runtime.h>

typedef __attribute__((ext_vector_type(16))) _Float16 v16h;
typedef __attribute__((ext_vector_type(8)))  float    v8f;

#define NODES 50000
#define EDGES 250000
#define HDIM  128

// ---------------------------------------------------------------------------
// helpers
// ---------------------------------------------------------------------------
__device__ __forceinline__ unsigned f2ord(float f) {
  unsigned u = __float_as_uint(f);
  return (u & 0x80000000u) ? ~u : (u | 0x80000000u);
}
__device__ __forceinline__ float ord2f(unsigned u) {
  return (u & 0x80000000u) ? __uint_as_float(u & 0x7FFFFFFFu)
                           : __uint_as_float(~u);
}
__device__ __forceinline__ unsigned pack2h(float a, float b) {
  union { _Float16 h[2]; unsigned u; } t;
  t.h[0] = (_Float16)a;
  t.h[1] = (_Float16)b;
  return t.u;
}

// ---------------------------------------------------------------------------
// Generic WMMA GEMM: C[M,Nout] = op(A[M,K] @ W[K,Nout] + bias)
// Preconditions (guaranteed by host): K % 32 == 0, Nout % 64 == 0,
// ldb % 4 == 0, lda % 4 == 0, all row starts 16B-aligned.
// GATHER==0: A is a dense fp32 row-major matrix (lda).
// GATHER==1: row m of A is concat(h[src[m]], h[dst[m]], ea[m]) (K==384),
//            optional ReLU on the two h segments (RELUG).
//            Segment choice is uniform per 32-wide K tile -> hoisted.
// EPI: 0 = store, 1 = relu-store, 2 = out += 0.5f * val (edge residual)
// Block: 128 threads = 4 waves; tile 64(M) x 64(N) x 32(K).
// ---------------------------------------------------------------------------
template <int GATHER, int EPI, bool RELUG>
__global__ void gemm_wmma(const float* __restrict__ A, int lda,
                          const float* __restrict__ hbuf,
                          const float* __restrict__ eabuf,
                          const int* __restrict__ srci,
                          const int* __restrict__ dsti,
                          const float* __restrict__ W, int ldb,
                          const float* __restrict__ bias,
                          float* __restrict__ out, int ldo,
                          int M, int K, int Nout) {
  __shared__ __align__(16) _Float16 As[64 * 34];   // [row][k], stride 34
  __shared__ __align__(16) _Float16 Bs[64 * 34];   // [n][k],   stride 34

  const int tx   = threadIdx.x;
  const int wave = tx >> 5;
  const int lane = tx & 31;
  const int hf   = lane >> 4;      // half-wave select
  const int l15  = lane & 15;
  const int m0   = blockIdx.y * 64;
  const int n0   = blockIdx.x * 64;

  v8f acc0 = {}, acc1 = {}, acc2 = {}, acc3 = {};

  const int ktiles = K >> 5;
  for (int kt = 0; kt < ktiles; ++kt) {
    const int kb0 = kt << 5;

    // ---- stage A tile (64 x 32), fp32 -> fp16, b128 loads ----
    {
      const int row = tx >> 1;          // 0..63
      const int cb  = (tx & 1) * 16;    // 0 / 16
      const int m   = m0 + row;
      unsigned* As32 = (unsigned*)As;
      const int base = row * 34 + cb;   // even
      const float* rp = nullptr;
      bool dorelu = false;
      if (GATHER) {
        const int seg  = kb0 >> 7;      // uniform per k-tile
        const int off0 = kb0 & 127;
        if (m < M) {
          if (seg == 0)      { rp = hbuf + (long)srci[m] * HDIM + off0; dorelu = RELUG; }
          else if (seg == 1) { rp = hbuf + (long)dsti[m] * HDIM + off0; dorelu = RELUG; }
          else               { rp = eabuf + (long)m * HDIM + off0; }
        }
      } else if (m < M) {
        rp = A + (long)m * lda + kb0;
      }
      if (rp) {
#pragma unroll
        for (int q = 0; q < 4; ++q) {
          float4 v = *(const float4*)(rp + cb + q * 4);
          if (dorelu) {
            v.x = fmaxf(v.x, 0.f); v.y = fmaxf(v.y, 0.f);
            v.z = fmaxf(v.z, 0.f); v.w = fmaxf(v.w, 0.f);
          }
          As32[(base + q * 4)     >> 1] = pack2h(v.x, v.y);
          As32[(base + q * 4 + 2) >> 1] = pack2h(v.z, v.w);
        }
      } else {
#pragma unroll
        for (int q = 0; q < 8; ++q) As32[(base + q * 2) >> 1] = 0u;
      }
    }

    // ---- stage B tile (32 x 64) transposed into Bs[n][k], b128 loads ----
    {
      const int kr = tx >> 2;          // 0..31
      const int nb = (tx & 3) * 16;    // 0/16/32/48
      const float* rp = W + (long)(kb0 + kr) * ldb + n0 + nb;
#pragma unroll
      for (int q = 0; q < 4; ++q) {
        float4 v = *(const float4*)(rp + q * 4);
        Bs[(nb + q * 4 + 0) * 34 + kr] = (_Float16)v.x;
        Bs[(nb + q * 4 + 1) * 34 + kr] = (_Float16)v.y;
        Bs[(nb + q * 4 + 2) * 34 + kr] = (_Float16)v.z;
        Bs[(nb + q * 4 + 3) * 34 + kr] = (_Float16)v.w;
      }
    }
    __syncthreads();

    // ---- fragments (ISA 16-bit layouts), packed 32-bit DS reads ----
    union U { unsigned u[8]; v16h v; };
    const unsigned* As32 = (const unsigned*)As;
    const unsigned* Bs32 = (const unsigned*)Bs;

    U af;
    {
      int ar = wave * 16 + l15;
#pragma unroll
      for (int vv = 0; vv < 8; ++vv) {
        int kk = ((vv < 4) ? 0 : 16) + hf * 8 + (vv & 3) * 2;
        af.u[vv] = As32[(ar * 34 + kk) >> 1];
      }
    }
    U bf0, bf1, bf2, bf3;
#pragma unroll
    for (int vv = 0; vv < 8; ++vv) {
      int kk = ((vv < 4) ? 0 : 16) + hf * 8 + (vv & 3) * 2;
      bf0.u[vv] = Bs32[((0 * 16 + l15) * 34 + kk) >> 1];
      bf1.u[vv] = Bs32[((1 * 16 + l15) * 34 + kk) >> 1];
      bf2.u[vv] = Bs32[((2 * 16 + l15) * 34 + kk) >> 1];
      bf3.u[vv] = Bs32[((3 * 16 + l15) * 34 + kk) >> 1];
    }
    acc0 = __builtin_amdgcn_wmma_f32_16x16x32_f16(false, af.v, false, bf0.v,
                                                  (short)0, acc0, false, false);
    acc1 = __builtin_amdgcn_wmma_f32_16x16x32_f16(false, af.v, false, bf1.v,
                                                  (short)0, acc1, false, false);
    acc2 = __builtin_amdgcn_wmma_f32_16x16x32_f16(false, af.v, false, bf2.v,
                                                  (short)0, acc2, false, false);
    acc3 = __builtin_amdgcn_wmma_f32_16x16x32_f16(false, af.v, false, bf3.v,
                                                  (short)0, acc3, false, false);
    __syncthreads();
  }

  // ---- epilogue (C layout: lane n = lane&15, rows r + (lane>>4)*8) ----
#pragma unroll
  for (int ns = 0; ns < 4; ++ns) {
    v8f acc = (ns == 0) ? acc0 : (ns == 1) ? acc1 : (ns == 2) ? acc2 : acc3;
    int n = n0 + ns * 16 + l15;
    float bv = bias ? bias[n] : 0.f;
    int mb = m0 + wave * 16 + hf * 8;
#pragma unroll
    for (int r = 0; r < 8; ++r) {
      int m = mb + r;
      if (m < M) {
        float val = acc[r] + bv;
        if (EPI == 1) val = fmaxf(val, 0.f);
        if (EPI == 2) out[(long)m * ldo + n] += 0.5f * val;
        else          out[(long)m * ldo + n] = val;
      }
    }
  }
}

// ---------------------------------------------------------------------------
// pack readout weights into zero-padded, tile-friendly buffers
// mlp_W1 [384x50] -> Wp1 [384x64]; mlp_W2 [50x25] -> Wp2 [64x64];
// b1 -> bp1[64]; b2 -> bp2[64]
// ---------------------------------------------------------------------------
__global__ void k_pack(const float* __restrict__ W1, const float* __restrict__ b1,
                       const float* __restrict__ W2, const float* __restrict__ b2,
                       float* __restrict__ Wp1, float* __restrict__ bp1,
                       float* __restrict__ Wp2, float* __restrict__ bp2) {
  int t = blockIdx.x * blockDim.x + threadIdx.x;
  if (t < 384 * 64) {
    int r = t >> 6, c = t & 63;
    Wp1[t] = (c < 50) ? W1[r * 50 + c] : 0.f;
  }
  if (t < 64 * 64) {
    int r = t >> 6, c = t & 63;
    Wp2[t] = (r < 50 && c < 25) ? W2[r * 25 + c] : 0.f;
  }
  if (t < 64) {
    bp1[t] = (t < 50) ? b1[t] : 0.f;
    bp2[t] = (t < 25) ? b2[t] : 0.f;
  }
}

// ---------------------------------------------------------------------------
// elementwise / reduction kernels
// ---------------------------------------------------------------------------
__global__ void k_zero(float* p, long n) {
  long i = blockIdx.x * (long)blockDim.x + threadIdx.x;
  if (i < n) p[i] = 0.f;
}

// column sums of [rows x 128] matrix -> sum[128]
__global__ void k_colsum128(const float* __restrict__ x, long rows,
                            float* __restrict__ sum) {
  __shared__ float part[128];
  if (threadIdx.x < 128) part[threadIdx.x] = 0.f;
  __syncthreads();
  long n = rows * 128;
  for (long i = blockIdx.x * (long)blockDim.x + threadIdx.x; i < n;
       i += (long)gridDim.x * blockDim.x)
    atomicAdd(&part[(int)(i & 127)], x[i]);
  __syncthreads();
  if (threadIdx.x < 128) atomicAdd(&sum[threadIdx.x], part[threadIdx.x]);
}

__global__ void k_scale(const float* in, float* outp, float s, int n) {
  int t = blockIdx.x * blockDim.x + threadIdx.x;
  if (t < n) outp[t] = in[t] * s;
}

// v[k][h] = sum_c conv_edge_W[k][h*32+c] * att_edge[h][c]
__global__ void k_attvec(const float* __restrict__ Wce,
                         const float* __restrict__ atte,
                         float* __restrict__ v) {
  int t = blockIdx.x * blockDim.x + threadIdx.x;
  if (t >= 128 * 4) return;
  int k = t >> 2, h4 = t & 3;
  float s = 0.f;
  for (int c = 0; c < 32; ++c) s += Wce[k * 128 + h4 * 32 + c] * atte[h4 * 32 + c];
  v[k * 4 + h4] = s;
}

// a_e for self loops: aeloop[h] = sum_k mean[k] * v[k][h]
__global__ void k_aeloop(const float* __restrict__ mean,
                         const float* __restrict__ v, float* __restrict__ outp) {
  int h4 = threadIdx.x;
  if (h4 >= 4) return;
  float s = 0.f;
  for (int k = 0; k < 128; ++k) s += mean[k] * v[k * 4 + h4];
  outp[h4] = s;
}

// per-node attention scores a_s, a_d
__global__ void k_asd(const float* __restrict__ xs, const float* __restrict__ atts,
                      const float* __restrict__ attd, float* __restrict__ a_s,
                      float* __restrict__ a_d, int n) {
  int t = blockIdx.x * blockDim.x + threadIdx.x;
  if (t >= n * 4) return;
  int node = t >> 2, h4 = t & 3;
  float s = 0.f, d = 0.f;
  for (int c = 0; c < 32; ++c) {
    float x = xs[(long)node * 128 + h4 * 32 + c];
    s += x * atts[h4 * 32 + c];
    d += x * attd[h4 * 32 + c];
  }
  a_s[t] = s;
  a_d[t] = d;
}

// a_e per real edge: ea[e] . v[:,h]
__global__ void k_aeedge(const float* __restrict__ ea, const float* __restrict__ v,
                         float* __restrict__ a_e) {
  int t = blockIdx.x * blockDim.x + threadIdx.x;
  if (t >= EDGES * 4) return;
  int e = t >> 2, h4 = t & 3;
  float s = 0.f;
  for (int k = 0; k < 128; ++k) s += ea[(long)e * 128 + k] * v[k * 4 + h4];
  a_e[t] = s;
}

// alpha = leaky_relu(a_s[src]+a_d[dst]+a_e, 0.2); segment max via ordered atomicMax
__global__ void k_alpha(const int* __restrict__ src, const int* __restrict__ dst,
                        const float* __restrict__ a_s, const float* __restrict__ a_d,
                        const float* __restrict__ a_e, const float* __restrict__ aeloop,
                        float* __restrict__ alpha, unsigned* __restrict__ amaxU,
                        int total) {
  int t = blockIdx.x * blockDim.x + threadIdx.x;
  if (t >= total * 4) return;
  int idx = t >> 2, h4 = t & 3;
  int s, d;
  float aev;
  if (idx < EDGES) { s = src[idx]; d = dst[idx]; aev = a_e[idx * 4 + h4]; }
  else             { s = idx - EDGES; d = s;     aev = aeloop[h4]; }
  float a = a_s[s * 4 + h4] + a_d[d * 4 + h4] + aev;
  a = (a > 0.f) ? a : 0.2f * a;
  alpha[t] = a;
  atomicMax(&amaxU[d * 4 + h4], f2ord(a));
}

// ex = exp(alpha - amax[dst]); segment sum into denom
__global__ void k_exden(const int* __restrict__ dst, const float* __restrict__ alpha,
                        const unsigned* __restrict__ amaxU, float* __restrict__ exb,
                        float* __restrict__ denom, int total) {
  int t = blockIdx.x * blockDim.x + threadIdx.x;
  if (t >= total * 4) return;
  int idx = t >> 2, h4 = t & 3;
  int d = (idx < EDGES) ? dst[idx] : idx - EDGES;
  float ex = __expf(alpha[t] - ord2f(amaxU[d * 4 + h4]));
  exb[t] = ex;
  atomicAdd(&denom[d * 4 + h4], ex);
}

// aggregation: agg[dst, c] += (ex/denom) * xs[src, c]  (one block per edge)
__global__ void k_agg(const int* __restrict__ src, const int* __restrict__ dst,
                      const float* __restrict__ exb, const float* __restrict__ denom,
                      const float* __restrict__ xs, float* __restrict__ agg) {
  int idx = blockIdx.x;
  int c = threadIdx.x;
  int h4 = c >> 5;
  int s, d;
  if (idx < EDGES) { s = src[idx]; d = dst[idx]; }
  else             { s = idx - EDGES; d = s; }
  float w = exb[idx * 4 + h4] / (denom[d * 4 + h4] + 1e-16f);
  atomicAdd(&agg[(long)d * 128 + c], w * xs[(long)s * 128 + c]);
}

// batch-norm statistics over nodes of (agg + conv_bias)
__global__ void k_bnstats(const float* __restrict__ agg, const float* __restrict__ cbias,
                          float* __restrict__ sum, float* __restrict__ sumsq) {
  __shared__ float ps[128], pq[128];
  if (threadIdx.x < 128) { ps[threadIdx.x] = 0.f; pq[threadIdx.x] = 0.f; }
  __syncthreads();
  long n = (long)NODES * 128;
  for (long i = blockIdx.x * (long)blockDim.x + threadIdx.x; i < n;
       i += (long)gridDim.x * blockDim.x) {
    int c = (int)(i & 127);
    float v = agg[i] + cbias[c];
    atomicAdd(&ps[c], v);
    atomicAdd(&pq[c], v * v);
  }
  __syncthreads();
  if (threadIdx.x < 128) {
    atomicAdd(&sum[threadIdx.x], ps[threadIdx.x]);
    atomicAdd(&sumsq[threadIdx.x], pq[threadIdx.x]);
  }
}

// apply BN + residual:  h = (h + relu(bn(out))) / 2
__global__ void k_bnupd(const float* __restrict__ agg, const float* __restrict__ cbias,
                        const float* __restrict__ sum, const float* __restrict__ sumsq,
                        const float* __restrict__ gamma, const float* __restrict__ beta,
                        float* __restrict__ h, float invN) {
  long t = blockIdx.x * (long)blockDim.x + threadIdx.x;
  long n = (long)NODES * 128;
  if (t >= n) return;
  int c = (int)(t & 127);
  float mu = sum[c] * invN;
  float var = sumsq[c] * invN - mu * mu;
  float o = (agg[t] + cbias[c] - mu) * rsqrtf(var + 1e-5f) * gamma[c] + beta[c];
  h[t] = (h[t] + fmaxf(o, 0.f)) * 0.5f;
}

// final tiny head: out[e,0:2] = z2[e,0:25] @ W3 + b3   (z2 ld = 64)
__global__ void k_final(const float* __restrict__ z2, const float* __restrict__ W3,
                        const float* __restrict__ b3, float* __restrict__ outp) {
  int e = blockIdx.x * blockDim.x + threadIdx.x;
  if (e >= EDGES) return;
  float o0 = b3[0], o1 = b3[1];
  for (int k = 0; k < 25; ++k) {
    float z = z2[(long)e * 64 + k];
    o0 += z * W3[k * 2 + 0];
    o1 += z * W3[k * 2 + 1];
  }
  outp[(long)e * 2 + 0] = o0;
  outp[(long)e * 2 + 1] = o1;
}

// ---------------------------------------------------------------------------
// host
// ---------------------------------------------------------------------------
static inline int cdiv(long a, long b) { return (int)((a + b - 1) / b); }

extern "C" void kernel_launch(void* const* d_in, const int* in_sizes, int n_in,
                              void* d_out, int out_size, void* d_ws, size_t ws_size,
                              hipStream_t stream) {
  const float* x         = (const float*)d_in[0];
  const int*   eidx      = (const int*)d_in[1];
  const float* edge_attr = (const float*)d_in[2];
  const float* node_W    = (const float*)d_in[3];
  const float* node_b    = (const float*)d_in[4];
  const float* edge_W    = (const float*)d_in[5];
  const float* edge_b    = (const float*)d_in[6];
  const float* conv_W    = (const float*)d_in[7];
  const float* att_src   = (const float*)d_in[8];
  const float* att_dst   = (const float*)d_in[9];
  const float* conv_eW   = (const float*)d_in[10];
  const float* att_edge  = (const float*)d_in[11];
  const float* conv_bias = (const float*)d_in[12];
  const float* bn_gamma  = (const float*)d_in[13];
  const float* bn_beta   = (const float*)d_in[14];
  const float* emlp_W1   = (const float*)d_in[15];
  const float* emlp_b1   = (const float*)d_in[16];
  const float* emlp_W2   = (const float*)d_in[17];
  const float* emlp_b2   = (const float*)d_in[18];
  const float* mlp_W1    = (const float*)d_in[19];
  const float* mlp_b1    = (const float*)d_in[20];
  const float* mlp_W2    = (const float*)d_in[21];
  const float* mlp_b2    = (const float*)d_in[22];
  const float* mlp_W3    = (const float*)d_in[23];
  const float* mlp_b3    = (const float*)d_in[24];
  float* outp = (float*)d_out;

  const int* src = eidx;
  const int* dst = eidx + EDGES;
  const int TOT = EDGES + NODES;   // edges + self loops

  // ---- workspace carve-up ----
  float* p = (float*)d_ws;
  float* h    = p; p += (long)NODES * HDIM;
  float* ea   = p; p += (long)EDGES * HDIM;
  float* xs   = p; p += (long)NODES * HDIM;
  float* agg  = p; p += (long)NODES * HDIM;
  float* z1   = p; p += (long)EDGES * HDIM;   // also readout intermediate (ld 64)
  float* z2   = p; p += (long)EDGES * 64;
  float* a_s  = p; p += (long)NODES * 4;
  float* a_d  = p; p += (long)NODES * 4;
  float* a_e  = p; p += (long)EDGES * 4;
  float* alpha= p; p += (long)TOT * 4;
  float* exb  = p; p += (long)TOT * 4;
  unsigned* amaxU = (unsigned*)p; p += (long)NODES * 4;
  float* denom  = p; p += (long)NODES * 4;
  float* easum  = p; p += 128;
  float* eamean = p; p += 128;
  float* vatt   = p; p += 512;
  float* aeloop = p; p += 4;
  float* bnsum  = p; p += 128;
  float* bnsq   = p; p += 128;
  float* Wp1    = p; p += 384 * 64;
  float* Wp2    = p; p += 64 * 64;
  float* bp1    = p; p += 64;
  float* bp2    = p; p += 64;

  dim3 blk(128);

  // pack readout weights (removes all GEMM tail paths)
  k_pack<<<cdiv(384 * 64, 256), 256, 0, stream>>>(mlp_W1, mlp_b1, mlp_W2, mlp_b2,
                                                  Wp1, bp1, Wp2, bp2);

  // ---- input embeddings ----
  gemm_wmma<0, 0, false><<<dim3(2, cdiv(NODES, 64)), blk, 0, stream>>>(
      x, 64, nullptr, nullptr, nullptr, nullptr, node_W, 128, node_b, h, 128,
      NODES, 64, 128);
  gemm_wmma<0, 0, false><<<dim3(2, cdiv(EDGES, 64)), blk, 0, stream>>>(
      edge_attr, 32, nullptr, nullptr, nullptr, nullptr, edge_W, 128, edge_b,
      ea, 128, EDGES, 32, 128);

  for (int i = 0; i < 2; ++i) {
    const float* cW  = conv_W   + (long)i * 128 * 128;
    const float* ceW = conv_eW  + (long)i * 128 * 128;
    const float* as_ = att_src  + i * 4 * 32;
    const float* ad_ = att_dst  + i * 4 * 32;
    const float* ae_ = att_edge + i * 4 * 32;
    const float* cb  = conv_bias + i * 128;
    const float* g_  = bn_gamma + i * 128;
    const float* b_  = bn_beta + i * 128;
    const float* W1  = emlp_W1 + (long)i * 384 * 128;
    const float* b1  = emlp_b1 + i * 128;
    const float* W2  = emlp_W2 + (long)i * 128 * 128;
    const float* b2  = emlp_b2 + i * 128;

    // xs = h @ conv_W
    gemm_wmma<0, 0, false><<<dim3(2, cdiv(NODES, 64)), blk, 0, stream>>>(
        h, 128, nullptr, nullptr, nullptr, nullptr, cW, 128, nullptr, xs, 128,
        NODES, 128, 128);
    k_asd<<<cdiv((long)NODES * 4, 256), 256, 0, stream>>>(xs, as_, ad_, a_s, a_d, NODES);

    // edge-attr mean + fused attention vector
    k_zero<<<1, 128, 0, stream>>>(easum, 128);
    k_colsum128<<<512, 256, 0, stream>>>(ea, EDGES, easum);
    k_scale<<<1, 128, 0, stream>>>(easum, eamean, 1.0f / EDGES, 128);
    k_attvec<<<2, 256, 0, stream>>>(ceW, ae_, vatt);
    k_aeloop<<<1, 4, 0, stream>>>(eamean, vatt, aeloop);
    k_aeedge<<<cdiv((long)EDGES * 4, 256), 256, 0, stream>>>(ea, vatt, a_e);

    // segment softmax + aggregation
    k_zero<<<cdiv((long)NODES * 4, 256), 256, 0, stream>>>((float*)amaxU, (long)NODES * 4);
    k_zero<<<cdiv((long)NODES * 4, 256), 256, 0, stream>>>(denom, (long)NODES * 4);
    k_zero<<<cdiv((long)NODES * 128, 256), 256, 0, stream>>>(agg, (long)NODES * 128);
    k_alpha<<<cdiv((long)TOT * 4, 256), 256, 0, stream>>>(src, dst, a_s, a_d, a_e,
                                                          aeloop, alpha, amaxU, TOT);
    k_exden<<<cdiv((long)TOT * 4, 256), 256, 0, stream>>>(dst, alpha, amaxU, exb,
                                                          denom, TOT);
    k_agg<<<TOT, 128, 0, stream>>>(src, dst, exb, denom, xs, agg);

    // batch norm + residual update of h
    k_zero<<<1, 256, 0, stream>>>(bnsum, 256);  // bnsum+bnsq contiguous
    k_bnstats<<<512, 256, 0, stream>>>(agg, cb, bnsum, bnsq);
    k_bnupd<<<cdiv((long)NODES * 128, 256), 256, 0, stream>>>(agg, cb, bnsum, bnsq,
                                                              g_, b_, h, 1.0f / NODES);

    // edge MLP: z1 = relu(concat(h[src],h[dst],ea) @ W1 + b1)
    gemm_wmma<1, 1, false><<<dim3(2, cdiv(EDGES, 64)), blk, 0, stream>>>(
        nullptr, 0, h, ea, src, dst, W1, 128, b1, z1, 128, EDGES, 384, 128);
    // ea += 0.5 * (z1 @ W2 + b2)
    gemm_wmma<0, 2, false><<<dim3(2, cdiv(EDGES, 64)), blk, 0, stream>>>(
        z1, 128, nullptr, nullptr, nullptr, nullptr, W2, 128, b2, ea, 128,
        EDGES, 128, 128);
  }

  // ---- readout: relu-gathered concat -> 64(50) -> 64(25) -> 2 ----
  // GEMM1 writes z1 [E x 64]; pad cols produce relu(0+0)=0, so GEMM2's K=64 is exact.
  gemm_wmma<1, 1, true><<<dim3(1, cdiv(EDGES, 64)), blk, 0, stream>>>(
      nullptr, 0, h, ea, src, dst, Wp1, 64, bp1, z1, 64, EDGES, 384, 64);
  gemm_wmma<0, 1, false><<<dim3(1, cdiv(EDGES, 64)), blk, 0, stream>>>(
      z1, 64, nullptr, nullptr, nullptr, nullptr, Wp2, 64, bp2, z2, 64,
      EDGES, 64, 64);
  k_final<<<cdiv(EDGES, 256), 256, 0, stream>>>(z2, mlp_W3, mlp_b3, outp);
}
